// DKD_88210038325779
// MI455X (gfx1250) — matrix operand order repaired and checked by reading, not compiled, not run on hardware
//
#include <hip/hip_runtime.h>
#include <math.h>

// ---------------- problem constants (from reference) ----------------
#define RADIUS 2
#define TOPK   500
#define NB     8
#define H      480
#define W      640
#define HW     (H*W)
#define NPIX   (NB*HW)
#define CAP    65536          // candidate capacity per batch (NMS survivors ~12-30k)
#define EPSV   1e-12f

// output layout (flat, in tuple return order)
#define OFF_KXY   0           // (8,500,2)  = 8000
#define OFF_DESC  8000        // (8,500,64) = 256000
#define OFF_SCORE 264000      // (8,500)    = 4000
#define OFF_DISP  268000      // (8,500)    = 4000

typedef __attribute__((ext_vector_type(16))) _Float16 v16h;
typedef __attribute__((ext_vector_type(8)))  float    v8f;

// 5x5 window max with clamped bounds (== -inf padded reduce_window)
__device__ __forceinline__ float win_max_f(const float* p, int y, int x) {
    float mx = -3.4e38f;
    int ylo = max(y - RADIUS, 0), yhi = min(y + RADIUS, H - 1);
    int xlo = max(x - RADIUS, 0), xhi = min(x + RADIUS, W - 1);
    for (int yy = ylo; yy <= yhi; ++yy)
        for (int xx = xlo; xx <= xhi; ++xx)
            mx = fmaxf(mx, p[yy * W + xx]);
    return mx;
}

// ---------------- NMS passes ----------------
__global__ void k_mask(const float* __restrict__ scores, unsigned char* __restrict__ mask) {
    int i = blockIdx.x * blockDim.x + threadIdx.x;
    if (i >= NPIX) return;
    int bi = i / HW, rem = i % HW, y = rem / W, x = rem % W;
    const float* sp = scores + (size_t)bi * HW;
    mask[i] = (sp[rem] == win_max_f(sp, y, x)) ? 1 : 0;
}

__global__ void k_supp(const float* __restrict__ scores, const unsigned char* __restrict__ mask,
                       unsigned char* __restrict__ suppmask, float* __restrict__ suppscores) {
    int i = blockIdx.x * blockDim.x + threadIdx.x;
    if (i >= NPIX) return;
    int bi = i / HW, rem = i % HW, y = rem / W, x = rem % W;
    const unsigned char* mp = mask + (size_t)bi * HW;
    unsigned char any = 0;
    int ylo = max(y - RADIUS, 0), yhi = min(y + RADIUS, H - 1);
    int xlo = max(x - RADIUS, 0), xhi = min(x + RADIUS, W - 1);
    for (int yy = ylo; yy <= yhi; ++yy)
        for (int xx = xlo; xx <= xhi; ++xx)
            any |= mp[yy * W + xx];
    suppmask[i]   = any;
    suppscores[i] = any ? 0.f : scores[i];
}

__global__ void k_update(const float* __restrict__ suppscores, const unsigned char* __restrict__ suppmask,
                         unsigned char* __restrict__ mask) {
    int i = blockIdx.x * blockDim.x + threadIdx.x;
    if (i >= NPIX) return;
    int bi = i / HW, rem = i % HW, y = rem / W, x = rem % W;
    const float* sp = suppscores + (size_t)bi * HW;
    if (!suppmask[i] && sp[rem] == win_max_f(sp, y, x)) mask[i] = 1;
}

// ---------------- candidate compaction + sort (top-k) ----------------
__global__ void k_zero(unsigned long long* __restrict__ keys, unsigned int* __restrict__ cnt) {
    int i = blockIdx.x * blockDim.x + threadIdx.x;
    if (i < NB * CAP) keys[i] = 0ULL;
    if (i < NB) cnt[i] = 0u;
}

__global__ void k_compact(const float* __restrict__ scores, const unsigned char* __restrict__ mask,
                          unsigned long long* __restrict__ keys, unsigned int* __restrict__ cnt) {
    int i = blockIdx.x * blockDim.x + threadIdx.x;
    if (i >= NPIX) return;
    if (!mask[i]) return;
    int bi = i / HW, rem = i % HW, y = rem / W, x = rem % W;
    // border zeroing: rows/cols 0..r and last r zeroed
    if (y <= RADIUS || y >= H - RADIUS || x <= RADIUS || x >= W - RADIUS) return;
    unsigned int pos = atomicAdd(&cnt[bi], 1u);
    if (pos < CAP) {
        // descending value, then ascending index (matches lax.top_k tie-break)
        unsigned long long key = ((unsigned long long)__float_as_uint(scores[i]) << 32)
                               | (unsigned long long)(0xFFFFFFFFu - (unsigned int)rem);
        keys[(size_t)bi * CAP + pos] = key;
    }
}

// one block per batch; full bitonic sort (descending) of next-pow2 >= count
__global__ void k_sort(unsigned long long* __restrict__ keysAll, const unsigned int* __restrict__ cnt) {
    unsigned long long* a = keysAll + (size_t)blockIdx.x * CAP;
    unsigned int c = cnt[blockIdx.x];
    if (c > CAP) c = CAP;
    int n = 512;
    while (n < (int)c) n <<= 1;          // zeros beyond count sort to the tail
    for (int k = 2; k <= n; k <<= 1) {
        for (int j = k >> 1; j > 0; j >>= 1) {
            for (int i = threadIdx.x; i < n; i += blockDim.x) {
                int ixj = i ^ j;
                if (ixj > i) {
                    unsigned long long ai = a[i], aj = a[ixj];
                    bool descBlock = ((i & k) == 0);
                    bool doSwap = descBlock ? (ai < aj) : (ai > aj);
                    if (doSwap) { a[i] = aj; a[ixj] = ai; }
                }
            }
            __syncthreads();
        }
    }
}

// ---------------- fused refine (WMMA) + sampling + descriptors ----------------
// one wave (32 lanes) per tile of 16 keypoints; 8 batches x 32 tiles = 256 blocks
__global__ void __launch_bounds__(32)
k_dkd(const float* __restrict__ scores, const float* __restrict__ desc_map,
      const unsigned long long* __restrict__ keysAll, float* __restrict__ out) {
    const int lane = threadIdx.x;
    const int tile = blockIdx.x & 31;
    const int bi   = blockIdx.x >> 5;
    const int kp0  = tile * 16;

    __shared__ float    Eraw[16][32];// raw patch staging (async global->LDS)
    __shared__ _Float16 E[16][32];   // exp patches: 16 keypoints x 32 taps (25 valid)
    __shared__ float    Mom[16][4];  // [Mx, My, M0, M2] per keypoint
    __shared__ int      sX[16], sY[16];
    __shared__ float    sPX[16], sPY[16];

    if (lane < 16) {
        int kp = kp0 + lane;
        unsigned long long key = (kp < TOPK) ? keysAll[(size_t)bi * CAP + kp] : 0ULL;
        unsigned int idx = 0xFFFFFFFFu - (unsigned int)(key & 0xFFFFFFFFu);
        if (idx >= (unsigned int)HW) idx = 0;   // padding safety
        sY[lane] = (int)(idx / W);
        sX[lane] = (int)(idx % W);
    }
    __syncthreads();

    const float* sp = scores + (size_t)bi * HW;

    // Phase 1: fire all 16x25 scattered patch loads as async global->LDS copies
    // (ASYNCcnt path; single wait below collapses 16 serialized L2 round-trips
    //  into one). OOB taps load a clamped-but-valid address; zeroed in phase 2.
    const int ty = lane / 5 - RADIUS;
    const int tx = lane % 5 - RADIUS;
    for (int m = 0; m < 16; ++m) {
        int yy = sY[m] + ty;
        int xx = sX[m] + tx;
        int yyc = min(max(yy, 0), H - 1);
        int xxc = min(max(xx, 0), W - 1);
        unsigned int lds_off = (unsigned int)(uintptr_t)(&Eraw[m][lane]);
        unsigned long long ga = (unsigned long long)(uintptr_t)(sp + yyc * W + xxc);
        asm volatile("global_load_async_to_lds_b32 %0, %1, off"
                     :: "v"(lds_off), "v"(ga) : "memory");
    }
    asm volatile("s_wait_asynccnt 0x0" ::: "memory");
    __syncthreads();

    // Phase 2: stabilized exp / T (T = 0.1) from the staged tile
    for (int m = 0; m < 16; ++m) {
        float v = -3.4e38f;
        if (lane < 25) {
            int yy = sY[m] + ty;
            int xx = sX[m] + tx;
            bool inb = (yy >= 0 && yy < H && xx >= 0 && xx < W);
            v = inb ? Eraw[m][lane] : 0.f;      // zero padding
        }
        float mv = v;
        for (int off = 16; off >= 1; off >>= 1) mv = fmaxf(mv, __shfl_xor(mv, off, 32));
        float e = (lane < 25) ? __expf((v - mv) * 10.f) : 0.f;
        E[m][lane] = (_Float16)e;
    }
    __syncthreads();

    // A: 16x32 f16 (ISA 7.12.2 layout: lane-half selects K chunk of 8)
    const int mrow = lane & 15;
    const int hi   = lane >> 4;
    v16h amat, bmat;
    for (int i2 = 0; i2 < 8; ++i2) {
        int kb = 2 * i2 + hi * 8 + ((i2 >= 4) ? 8 : 0);
        amat[2 * i2]     = E[mrow][kb];
        amat[2 * i2 + 1] = E[mrow][kb + 1];
    }
    // B: 32x16 f16, columns = [dx, dy, 1, dx^2+dy^2, 0...]; lane-half selects K chunk of 8
    const int ncol = lane & 15;
    for (int j = 0; j < 8; ++j) {
        int kb = 2 * j + hi * 8 + ((j >= 4) ? 8 : 0);
        for (int u = 0; u < 2; ++u) {
            int k = kb + u;
            float g = 0.f;
            if (k < 25) {
                float dx = (float)(k % 5 - RADIUS);
                float dy = (float)(k / 5 - RADIUS);
                g = (ncol == 0) ? dx : (ncol == 1) ? dy : (ncol == 2) ? 1.f
                  : (ncol == 3) ? (dx * dx + dy * dy) : 0.f;
            }
            bmat[2 * j + u] = (_Float16)g;
        }
    }
    v8f cacc = {};
    cacc = __builtin_amdgcn_wmma_f32_16x16x32_f16(false, amat, false, bmat,
                                                  (short)0, cacc, false, false);
    if (ncol < 4) {
        for (int j = 0; j < 8; ++j) Mom[j + hi * 8][ncol] = cacc[j];
    }
    __syncthreads();

    // moments -> sub-pixel offset, dispersity, kxy, score sample
    if (lane < 16) {
        int kp = kp0 + lane;
        float Mx = Mom[lane][0], My = Mom[lane][1], M0 = Mom[lane][2], M2 = Mom[lane][3];
        float inv = 1.f / (M0 + EPSV);
        float xr = Mx * inv, yr = My * inv;
        float disp = (M2 - 2.f * xr * Mx - 2.f * yr * My + (xr * xr + yr * yr) * M0)
                   * (1.f / (float)(RADIUS * RADIUS)) * inv;
        float fx = ((float)sX[lane] + xr) / (float)(W - 1) * 2.f - 1.f;
        float fy = ((float)sY[lane] + yr) / (float)(H - 1) * 2.f - 1.f;
        float px = fminf(fmaxf((float)sX[lane] + xr, 0.f), (float)(W - 1));
        float py = fminf(fmaxf((float)sY[lane] + yr, 0.f), (float)(H - 1));
        sPX[lane] = px; sPY[lane] = py;
        if (kp < TOPK) {
            int o = bi * TOPK + kp;
            out[OFF_KXY + 2 * o]     = fx;
            out[OFF_KXY + 2 * o + 1] = fy;
            out[OFF_DISP + o]        = disp;
            int x0 = (int)floorf(px), y0 = (int)floorf(py);
            float wx = px - (float)x0, wy = py - (float)y0;
            int x1 = min(x0 + 1, W - 1), y1 = min(y0 + 1, H - 1);
            float v00 = sp[y0 * W + x0], v01 = sp[y0 * W + x1];
            float v10 = sp[y1 * W + x0], v11 = sp[y1 * W + x1];
            out[OFF_SCORE + o] = v00 * (1.f - wx) * (1.f - wy) + v01 * wx * (1.f - wy)
                               + v10 * (1.f - wx) * wy + v11 * wx * wy;
        }
    }
    __syncthreads();

    // 64-channel descriptor bilinear sample + L2 normalize (2 channels per lane)
    for (int m = 0; m < 16; ++m) {
        int kp = kp0 + m;
        if (kp >= TOPK) continue;      // uniform across the wave
        float px = sPX[m], py = sPY[m];
        int x0 = (int)floorf(px), y0 = (int)floorf(py);
        float wx = px - (float)x0, wy = py - (float)y0;
        int x1 = min(x0 + 1, W - 1), y1 = min(y0 + 1, H - 1);
        float w00 = (1.f - wx) * (1.f - wy), w01 = wx * (1.f - wy);
        float w10 = (1.f - wx) * wy,         w11 = wx * wy;
        float val[2];
        float ss = 0.f;
        for (int p = 0; p < 2; ++p) {
            int ch = lane + 32 * p;
            const float* dp = desc_map + (size_t)(bi * 64 + ch) * HW;
            float v = w00 * dp[y0 * W + x0] + w01 * dp[y0 * W + x1]
                    + w10 * dp[y1 * W + x0] + w11 * dp[y1 * W + x1];
            val[p] = v;
            ss += v * v;
        }
        for (int off = 16; off >= 1; off >>= 1) ss += __shfl_xor(ss, off, 32);
        float invn = 1.f / fmaxf(sqrtf(ss), EPSV);
        int obase = OFF_DESC + (bi * TOPK + kp) * 64;
        out[obase + lane]      = val[0] * invn;
        out[obase + lane + 32] = val[1] * invn;
    }
}

// ---------------- launcher ----------------
extern "C" void kernel_launch(void* const* d_in, const int* in_sizes, int n_in,
                              void* d_out, int out_size, void* d_ws, size_t ws_size,
                              hipStream_t stream) {
    (void)in_sizes; (void)n_in; (void)out_size; (void)ws_size;
    const float* scores   = (const float*)d_in[0];
    const float* desc_map = (const float*)d_in[1];
    float* out = (float*)d_out;

    char* ws = (char*)d_ws;
    unsigned char* mask       = (unsigned char*)ws;                       // NPIX bytes
    unsigned char* suppmask   = (unsigned char*)(ws + (size_t)NPIX);      // NPIX bytes
    float* suppscores         = (float*)(ws + 2ull * NPIX);               // NPIX*4 bytes
    // keys overlap suppscores (disjoint lifetimes: suppscores dead after k_update)
    unsigned long long* keys  = (unsigned long long*)(ws + 2ull * NPIX);  // NB*CAP*8 <= NPIX*4
    unsigned int* cnt         = (unsigned int*)(ws + 2ull * NPIX + 4ull * NPIX);

    const int pixBlocks = (NPIX + 255) / 256;

    k_mask<<<pixBlocks, 256, 0, stream>>>(scores, mask);
    for (int it = 0; it < 2; ++it) {
        k_supp<<<pixBlocks, 256, 0, stream>>>(scores, mask, suppmask, suppscores);
        k_update<<<pixBlocks, 256, 0, stream>>>(suppscores, suppmask, mask);
    }
    k_zero<<<(NB * CAP + 255) / 256, 256, 0, stream>>>(keys, cnt);
    k_compact<<<pixBlocks, 256, 0, stream>>>(scores, mask, keys, cnt);
    k_sort<<<NB, 1024, 0, stream>>>(keys, cnt);
    k_dkd<<<NB * 32, 32, 0, stream>>>(scores, desc_map, keys, out);
}
